// StyledConv_61151744360934
// MI455X (gfx1250) — compile-verified, matrix operand order checked by reading, not run on hardware
//
#include <hip/hip_runtime.h>

typedef __attribute__((ext_vector_type(16))) __bf16 v16bf;
typedef __attribute__((ext_vector_type(8)))  __bf16 bf16x8;
typedef __attribute__((ext_vector_type(8)))  float  v8f;
typedef __attribute__((ext_vector_type(4)))  unsigned u32x4;
typedef __attribute__((ext_vector_type(4)))  int      i32x4;
typedef __attribute__((ext_vector_type(8)))  int      i32x8;

static constexpr int B_    = 8;
static constexpr int CIN_  = 512;
static constexpr int COUT_ = 512;
static constexpr int SDIM_ = 512;
static constexpr int H_    = 64;
static constexpr int W_    = 64;
static constexpr int HW_   = H_ * W_;
static constexpr int KTOT  = CIN_ * 9;   // 4608 = flattened (cin,kh,kw)
static constexpr int BK    = 32;         // one bf16 WMMA k-step
static constexpr int BM    = 128;        // co tile
static constexpr int BN    = 64;         // exactly one image row
static constexpr int AS    = BK + 8;     // padded LDS stride (elems): 80B rows
static constexpr int BS    = BK + 8;

// ---------- stage 1: s[b,cin] = style @ (modW^T / sqrt(SDIM)) + mod_bias ----------
__global__ void style_mod_kernel(const float* __restrict__ style,
                                 const float* __restrict__ mod_weight,
                                 const float* __restrict__ mod_bias,
                                 float* __restrict__ s_out) {
  int t  = blockIdx.x * blockDim.x + threadIdx.x;   // 0..4095
  int b  = t >> 9;
  int ci = t & 511;
  const float lin_scale = 0.04419417382415922f;     // 1/sqrt(512)
  const float* st = style + (size_t)b * SDIM_;
  const float* mw = mod_weight + (size_t)ci * SDIM_;
  float acc = 0.f;
  for (int d = 0; d < SDIM_; ++d) acc += st[d] * mw[d];
  s_out[t] = acc * lin_scale + mod_bias[ci];
}

// ---------- stage 2: wsq[co,cin] = sum_kk weight^2 ----------
__global__ void wsq_kernel(const float* __restrict__ w, float* __restrict__ wsq) {
  int t = blockIdx.x * blockDim.x + threadIdx.x;    // 0..262143
  const float* p = w + (size_t)t * 9;
  float a = 0.f;
#pragma unroll
  for (int i = 0; i < 9; ++i) { float v = p[i]; a += v * v; }
  wsq[t] = a;
}

// ---------- stage 3: demod[b,co] = rsqrt(wscale^2 * sum_cin wsq*s^2 + 1e-8) ----------
__global__ void demod_kernel(const float* __restrict__ wsq,
                             const float* __restrict__ s,
                             float* __restrict__ demod) {
  int t  = blockIdx.x * blockDim.x + threadIdx.x;   // 0..4095
  int b  = t >> 9;
  int co = t & 511;
  const float* sq = wsq + (size_t)co * CIN_;
  const float* sb = s   + (size_t)b  * CIN_;
  float a = 0.f;
  for (int ci = 0; ci < CIN_; ++ci) { float sv = sb[ci]; a += sq[ci] * sv * sv; }
  demod[t] = rsqrtf(a * (1.0f / 4608.0f) + 1e-8f);  // wscale^2 = 1/4608
}

// ---------- stage 4: weight f32 -> bf16 flat [COUT][KTOT] (batch-independent A) ----------
__global__ void wbf_kernel(const float* __restrict__ w, __bf16* __restrict__ wbf) {
  size_t base = ((size_t)blockIdx.x * blockDim.x + threadIdx.x) * 8;
  const float4 a = *(const float4*)(w + base);
  const float4 b = *(const float4*)(w + base + 4);
  bf16x8 o;
  o[0] = (__bf16)a.x; o[1] = (__bf16)a.y; o[2] = (__bf16)a.z; o[3] = (__bf16)a.w;
  o[4] = (__bf16)b.x; o[5] = (__bf16)b.y; o[6] = (__bf16)b.z; o[7] = (__bf16)b.w;
  *(bf16x8*)(wbf + base) = o;
}

// issue a TDM 2D tile load (bf16 weight tile -> padded LDS rows), per ISA D# layout
__device__ __forceinline__ void tdm_load_a(unsigned ldsA, unsigned long long ga) {
  u32x4 g0 = { 1u,                                        // count=1, user descriptor
               ldsA,                                      // lds_addr (bytes)
               (unsigned)ga,                              // global_addr[31:0]
               (unsigned)((ga >> 32) & 0x01FFFFFFu) | 0x80000000u };  // ga[56:32] | type=2
  i32x8 g1 = { (int)((1u << 16) |                         // data_size = 2B
                     (1u << 20) |                         // pad_enable
                     (3u << 22) |                         // pad_interval: 16 DWORDs (64B row)
                     (3u << 25)),                         // pad_amount: 4 DWORDs (16B pad)
               (int)((unsigned)(KTOT & 0xFFFF) << 16),    // tensor_dim0[15:0] (4608)
               (int)((unsigned)COUT_ << 16),              // tensor_dim1[15:0] (512)
               (int)((unsigned)BK << 16),                 // tile_dim0 = 32
               (int)BM,                                   // tile_dim1 = 128
               (int)KTOT,                                 // tensor_dim0_stride = 4608
               0, 0 };
  i32x4 z4 = { 0, 0, 0, 0 };
#if defined(__clang_major__) && (__clang_major__ >= 23)
  i32x8 z8 = { 0, 0, 0, 0, 0, 0, 0, 0 };
  __builtin_amdgcn_tensor_load_to_lds(g0, g1, z4, z4, z8, 0);
#else
  __builtin_amdgcn_tensor_load_to_lds(g0, g1, z4, z4, 0);
#endif
}

// ---------- stage 5: implicit-GEMM modulated conv, double-buffered, TDM + WMMA ----------
__global__ __launch_bounds__(256) void conv_wmma_kernel(
    const float* __restrict__ x, const __bf16* __restrict__ wbf,
    const float* __restrict__ s, const float* __restrict__ demod,
    const float* __restrict__ noise, const float* __restrict__ noise_weight,
    const float* __restrict__ bias, float* __restrict__ out) {
  const int h      = blockIdx.x;           // one image row per block (BN = W = 64)
  const int coBase = blockIdx.y * BM;
  const int b      = blockIdx.z;
  const int tid    = threadIdx.x;

  __shared__ __align__(16) __bf16 amLds[2][BM * AS];  // A: bf16 weight tiles (TDM-filled)
  __shared__ __align__(16) __bf16 bnLds[2][BN * BS];  // B: im2col x * smod, [n][k]
  __shared__ __align__(16) __bf16 smodBf[KTOT];       // bf16(wscale * s[b, k/9])
  __shared__ __align__(16) int    ktab[KTOT];         // (cin*HW+hh*W+kw-1)<<3 | kw<<1 | vh
  __shared__ float dmLds[BM];

  const float wscale = 0.014731391274719739f;         // 1/sqrt(4608)
  // ---- per-block setup tables (all integer divisions happen once, here)
  for (int k = tid; k < KTOT; k += 256) {
    int cin = (int)((unsigned)k / 9u);
    int kid = k - cin * 9;
    int kh  = kid / 3;
    int kw  = kid - kh * 3;
    smodBf[k] = (__bf16)(wscale * s[(size_t)b * CIN_ + cin]);
    int hh  = h + kh - 1;
    int vh  = ((unsigned)hh < (unsigned)H_) ? 1 : 0;
    int off = cin * HW_ + hh * W_ + (kw - 1);
    ktab[k] = (off << 3) | (kw << 1) | vh;
  }
  if (tid < BM) dmLds[tid] = demod[(size_t)b * COUT_ + coBase + tid];
  __syncthreads();

  const int lane = tid & 31;
  const int wv   = tid >> 5;
  const int wm   = wv & 3;                 // wave's 32-row strip of the 128 co rows
  const int wn   = wv >> 2;                // wave's 32-col strip of the 64 w cols
  const int l15  = lane & 15;
  const int koff = (lane < 16) ? 0 : 8;    // ISA 16-bit A/B striping

  v8f acc[2][2] = {};

  const int bn_n = tid & 63;               // B: one column (w) per thread
  const int bkb  = (tid >> 6) * 8;         // 8 consecutive k per thread
  // per-thread validity of ww = bn_n + kw - 1 for kw = 0,1,2
  const unsigned wwmask = (bn_n >= 1 ? 1u : 0u) | 2u | (bn_n <= W_ - 2 ? 4u : 0u);

  const float* xb = x + (size_t)b * CIN_ * HW_;
  const unsigned ldsA0 = (unsigned)(size_t)(&amLds[0][0]);
  const unsigned ldsA1 = (unsigned)(size_t)(&amLds[1][0]);
  const unsigned long long wbase =
      (unsigned long long)(size_t)wbf + (unsigned long long)coBase * KTOT * 2ull;

  // B-tile loader: vectorized table reads, im2col gather, scale, bf16 pack
  auto loadB = [&](int buf, int k0) {
    int tarr[8];
    *(i32x4*)&tarr[0] = *(const i32x4*)&ktab[k0 + bkb];
    *(i32x4*)&tarr[4] = *(const i32x4*)&ktab[k0 + bkb + 4];
    bf16x8 sm = *(const bf16x8*)&smodBf[k0 + bkb];
    bf16x8 o;
#pragma unroll
    for (int c = 0; c < 8; ++c) {
      int tt = tarr[c];
      int kw = (tt >> 1) & 3;
      bool ok = (tt & 1) && ((wwmask >> kw) & 1);
      float v = ok ? xb[(tt >> 3) + bn_n] : 0.f;
      o[c] = (__bf16)(v * (float)sm[c]);
    }
    *(bf16x8*)&bnLds[buf][bn_n * BS + bkb] = o;
  };

  // one 32-k WMMA step from LDS buffer `buf`
  auto computeTile = [&](int buf) {
    v16bf af[2], bfrag[2];
#pragma unroll
    for (int i = 0; i < 2; ++i) {
      int row = (wm * 2 + i) * 16 + l15;
      bf16x8 lo = *(const bf16x8*)&amLds[buf][row * AS + koff];
      bf16x8 hi = *(const bf16x8*)&amLds[buf][row * AS + koff + 16];
      af[i] = __builtin_shufflevector(lo, hi, 0,1,2,3,4,5,6,7,8,9,10,11,12,13,14,15);
    }
#pragma unroll
    for (int j = 0; j < 2; ++j) {
      int col = (wn * 2 + j) * 16 + l15;
      bf16x8 lo = *(const bf16x8*)&bnLds[buf][col * BS + koff];
      bf16x8 hi = *(const bf16x8*)&bnLds[buf][col * BS + koff + 16];
      bfrag[j] = __builtin_shufflevector(lo, hi, 0,1,2,3,4,5,6,7,8,9,10,11,12,13,14,15);
    }
#pragma unroll
    for (int i = 0; i < 2; ++i)
#pragma unroll
      for (int j = 0; j < 2; ++j)
        acc[i][j] = __builtin_amdgcn_wmma_f32_16x16x32_bf16(
            false, af[i], false, bfrag[j], (short)0, acc[i][j], false, false);
  };

  // ---- software pipeline: prefetch (TDM + B) into ping while computing pong
  if (wv == 0) tdm_load_a(ldsA0, wbase);
  loadB(0, 0);
  if (wv == 0) __builtin_amdgcn_s_wait_tensorcnt(0);
  __syncthreads();

  for (int k0 = 0; k0 < KTOT; k0 += 2 * BK) {   // 144 steps -> 72 double-steps
    int k1 = k0 + BK;                            // always valid (even step count)
    int k2 = k0 + 2 * BK;
    if (wv == 0) tdm_load_a(ldsA1, wbase + (unsigned long long)k1 * 2ull);
    loadB(1, k1);
    computeTile(0);
    if (wv == 0) __builtin_amdgcn_s_wait_tensorcnt(0);
    __syncthreads();
    if (k2 < KTOT) {
      if (wv == 0) tdm_load_a(ldsA0, wbase + (unsigned long long)k2 * 2ull);
      loadB(0, k2);
    }
    computeTile(1);
    if (wv == 0) __builtin_amdgcn_s_wait_tensorcnt(0);
    __syncthreads();
  }

  // ---- epilogue: *demod, + noise_weight*noise + bias, leaky_relu(0.2), * sqrt(2)
  const float nw = noise_weight[0];
  const float g  = 1.4142135623730951f;
#pragma unroll
  for (int i = 0; i < 2; ++i) {
#pragma unroll
    for (int j = 0; j < 2; ++j) {
      int wgl  = (wn * 2 + j) * 16 + l15;
      float nz = nw * noise[(size_t)b * HW_ + h * W_ + wgl];
#pragma unroll
      for (int r = 0; r < 8; ++r) {
        int mo  = (wm * 2 + i) * 16 + ((lane < 16) ? r : r + 8);
        int co  = coBase + mo;
        float v = acc[i][j][r] * dmLds[mo] + nz + bias[co];
        v = (v > 0.f) ? v : 0.2f * v;
        out[((size_t)b * COUT_ + co) * HW_ + h * W_ + wgl] = v * g;
      }
    }
  }
}

extern "C" void kernel_launch(void* const* d_in, const int* in_sizes, int n_in,
                              void* d_out, int out_size, void* d_ws, size_t ws_size,
                              hipStream_t stream) {
  (void)in_sizes; (void)n_in; (void)out_size; (void)ws_size;
  const float* x            = (const float*)d_in[0];
  const float* style        = (const float*)d_in[1];
  const float* weight       = (const float*)d_in[2];
  const float* mod_weight   = (const float*)d_in[3];
  const float* mod_bias     = (const float*)d_in[4];
  const float* noise        = (const float*)d_in[5];
  const float* noise_weight = (const float*)d_in[6];
  const float* bias         = (const float*)d_in[7];
  float* out = (float*)d_out;

  float*  ws        = (float*)d_ws;
  float*  s_buf     = ws;                 // 4096 floats
  float*  demod_buf = ws + 4096;          // 4096 floats
  float*  wsq_buf   = ws + 8192;          // 262144 floats
  __bf16* wbf       = (__bf16*)(ws + 8192 + 262144);  // 512*4608 bf16 (~4.7 MB)

  style_mod_kernel<<<16,   256, 0, stream>>>(style, mod_weight, mod_bias, s_buf);
  wsq_kernel      <<<1024, 256, 0, stream>>>(weight, wsq_buf);
  demod_kernel    <<<16,   256, 0, stream>>>(wsq_buf, s_buf, demod_buf);
  wbf_kernel      <<<1152, 256, 0, stream>>>(weight, wbf);   // 512*4608/8/256

  dim3 grid(H_, COUT_ / BM, B_);   // (64, 4, 8)
  conv_wmma_kernel<<<grid, 256, 0, stream>>>(x, wbf, s_buf, demod_buf,
                                             noise, noise_weight, bias, out);
}
